// TransducerLoss_28724741275703
// MI455X (gfx1250) — compile-verified, matrix operand order checked by reading, not run on hardware
//
#include <hip/hip_runtime.h>
#include <math.h>

// Problem constants (fixed by the reference).
#define B_ 32
#define T_ 256
#define U_ 128
#define V_ 4096

typedef __attribute__((ext_vector_type(16))) _Float16 v16h;
typedef __attribute__((ext_vector_type(8)))  float    v8f;

union FragU { v16h v; uint4 q[2]; };

// ---------------------------------------------------------------------------
// Pass 1: per-row max over V, then exp(x - max) stored as f16.
// One 256-thread block per row. Wave32 shuffle reduce + tiny LDS reduce.
// ---------------------------------------------------------------------------
__global__ __launch_bounds__(256)
void row_exp_kernel(const float* __restrict__ src, _Float16* __restrict__ dst,
                    float* __restrict__ rowmax) {
  const int row = blockIdx.x;
  const float* r = src + (size_t)row * V_;
  float m = -INFINITY;
  for (int v = threadIdx.x; v < V_; v += 256) m = fmaxf(m, r[v]);
#pragma unroll
  for (int off = 16; off > 0; off >>= 1) m = fmaxf(m, __shfl_xor(m, off, 32));
  __shared__ float red[8];
  __shared__ float bmax;
  if ((threadIdx.x & 31) == 0) red[threadIdx.x >> 5] = m;
  __syncthreads();
  if (threadIdx.x == 0) {
    float mm = red[0];
#pragma unroll
    for (int i = 1; i < 8; ++i) mm = fmaxf(mm, red[i]);
    bmax = mm;
    rowmax[row] = mm;
  }
  __syncthreads();
  m = bmax;
  _Float16* d = dst + (size_t)row * V_;
  for (int v = threadIdx.x; v < V_; v += 256) d[v] = (_Float16)__expf(r[v] - m);
}

// ---------------------------------------------------------------------------
// Pass 2 (fused): batched GEMM sum_v expE[b,t,v]*expP[b,u,v] via
// v_wmma_f32_16x16x32_f16, then the epilogue directly emits
//   blank_lp[b,t,u] = e[b,t,0] + p[b,u,0] - ln
//   lab_lp  [b,t,u] = e[b,t,lab] + p[b,u,lab] - ln   (u < U-1)
// with ln = log(acc) + maxE[t] + maxP[u].  No ln round-trip, no gather pass.
// BM=64, BN=128(=U), KC=32 -> grid (T/64, B) = 128 workgroups of 8 waves.
// Wave grid 4(M,16 rows) x 2(N,64 cols); 4 accum subtiles per wave.
// ---------------------------------------------------------------------------
__global__ __launch_bounds__(256)
void gemm_fused_kernel(const _Float16* __restrict__ expE,
                       const _Float16* __restrict__ expP,
                       const float* __restrict__ maxE,
                       const float* __restrict__ maxP,
                       const float* __restrict__ e,      // raw emissions (f32)
                       const float* __restrict__ p,      // raw predictions (f32)
                       const int*   __restrict__ labels,
                       float* __restrict__ blank_lp,
                       float* __restrict__ lab_lp) {
  const int b  = blockIdx.y;
  const int t0 = blockIdx.x * 64;

  // Padded LDS tiles (stride 40 halfs = 80B, 16B-aligned rows).
  __shared__ _Float16 lA[64][40];
  __shared__ _Float16 lB[128][40];

  const int tid  = threadIdx.x;
  const int lane = tid & 31;
  const int wid  = tid >> 5;
  const int wm   = wid & 3;   // 0..3: 16-row M subtile
  const int wn   = wid >> 2;  // 0..1: 64-col N band

  // 16-bit A/B fragment layout (ISA 7.12.2): lane<16 -> K 0..7 & 16..23,
  // lane>=16 -> K 8..15 & 24..31; row/col index = lane & 15.
  const int mrow  = lane & 15;
  const int khalf = (lane >> 4) << 3;  // 0 or 8

  v8f zero = {};
  v8f acc[4];
#pragma unroll
  for (int j = 0; j < 4; ++j) acc[j] = zero;

  const _Float16* Abase = expE + ((size_t)b * T_ + t0) * V_;
  const _Float16* Bbase = expP + ((size_t)b * U_) * V_;

  // A-tile: 64 rows x 4 16B-chunks = 256 chunks, 1 per thread.
  const int arow = tid >> 2;
  const int acol = (tid & 3) << 3;

  for (int kc = 0; kc < V_; kc += 32) {
    uint4 va = *(const uint4*)(Abase + (size_t)arow * V_ + kc + acol);
    *(uint4*)&lA[arow][acol] = va;
    // B-tile: 128 rows x 4 chunks = 512 chunks, 2 per thread.
#pragma unroll
    for (int s = 0; s < 2; ++s) {
      int c   = tid + s * 256;
      int row = c >> 2;
      int col = (c & 3) << 3;
      uint4 vb = *(const uint4*)(Bbase + (size_t)row * V_ + kc + col);
      *(uint4*)&lB[row][col] = vb;
    }
    __syncthreads();

    FragU fa, fb[4];
    {
      int r = wm * 16 + mrow;
      fa.q[0] = *(const uint4*)&lA[r][khalf];
      fa.q[1] = *(const uint4*)&lA[r][khalf + 16];
    }
#pragma unroll
    for (int j = 0; j < 4; ++j) {
      int r = wn * 64 + j * 16 + mrow;
      fb[j].q[0] = *(const uint4*)&lB[r][khalf];
      fb[j].q[1] = *(const uint4*)&lB[r][khalf + 16];
    }

#pragma unroll
    for (int j = 0; j < 4; ++j)
      acc[j] = __builtin_amdgcn_wmma_f32_16x16x32_f16(
          false, fa.v, false, fb[j].v, (short)0, acc[j], false, false);
    __syncthreads();
  }

  // Fused epilogue. f32 C/D layout: VGPR r -> M = r (+8 lanes>=16), N = lane&15.
  const int nlo = lane & 15;
  const int mhi = (lane >> 4) << 3;

  float me[8], e0[8];
  int   trow[8];
#pragma unroll
  for (int r = 0; r < 8; ++r) {
    int t  = t0 + wm * 16 + mhi + r;
    trow[r] = t;
    me[r] = maxE[b * T_ + t];
    e0[r] = e[((size_t)b * T_ + t) * V_];        // e[b,t,BLANK=0]
  }

#pragma unroll
  for (int j = 0; j < 4; ++j) {
    int n = wn * 64 + j * 16 + nlo;
    float mp = maxP[b * U_ + n];
    float p0 = p[((size_t)b * U_ + n) * V_];     // p[b,u,BLANK=0]
    bool  has_lab = (n < U_ - 1);
    int   lab = has_lab ? labels[b * (U_ - 1) + n] : 0;
    float pl  = has_lab ? p[((size_t)b * U_ + n) * V_ + lab] : 0.f;
#pragma unroll
    for (int r = 0; r < 8; ++r) {
      float lnv = __logf(acc[j][r]) + me[r] + mp;
      size_t idx = ((size_t)b * T_ + trow[r]) * U_ + n;
      blank_lp[idx] = e0[r] + p0 - lnv;
      if (has_lab) {
        float el = e[((size_t)b * T_ + trow[r]) * V_ + lab];
        lab_lp[idx] = el + pl - lnv;
      }
    }
  }
}

// ---------------------------------------------------------------------------
// Pass 3: RNN-T forward DP.  One wave32 per batch; lane owns u=4*lane..4*lane+3
// (float4 row loads).  Row update alpha = C + cumlogsumexp(a - C) with
// C = exclusive prefix-sum of the lab row.  All scans are barrier-free:
// 4-element local scan in registers + __shfl_up wave scan.
// ---------------------------------------------------------------------------
__device__ __forceinline__ float lse2(float a, float b) {
  float m = fmaxf(a, b);
  if (m == -INFINITY) return -INFINITY;
  return m + __logf(__expf(a - m) + __expf(b - m));
}

__global__ __launch_bounds__(32)
void dp_kernel(const float* __restrict__ blank_lp, const float* __restrict__ lab_lp,
               const int* __restrict__ Tls, const int* __restrict__ Uls,
               float* __restrict__ out) {
  const int b = blockIdx.x, lane = threadIdx.x;
  const int Tl = Tls[b], Ul = Uls[b];
  const float4* bl4 = (const float4*)(blank_lp + (size_t)b * T_ * U_);
  const float4* la4 = (const float4*)(lab_lp  + (size_t)b * T_ * U_);
  const int RS = U_ / 4;  // row stride in float4

  float a0, a1, a2, a3;   // alpha at u = 4*lane + {0,1,2,3}

  // t = 0 row: alpha = exclusive cumsum of lab row 0 (u=127 slot is not a label).
  {
    float4 lv = la4[lane];
    float l0 = lv.x, l1 = lv.y, l2 = lv.z, l3 = (lane == 31) ? 0.f : lv.w;
    float s1 = l0 + l1, s2 = s1 + l2, s3 = s2 + l3;
    float inc = s3;
#pragma unroll
    for (int off = 1; off < 32; off <<= 1) {
      float t = __shfl_up(inc, off, 32);
      if (lane >= off) inc += t;
    }
    float excl = inc - s3;
    a0 = excl; a1 = excl + l0; a2 = excl + s1; a3 = excl + s2;
  }

  for (int t = 1; t < Tl; ++t) {
    float4 bv = bl4[(size_t)(t - 1) * RS + lane];
    float4 lv = la4[(size_t)t * RS + lane];
    float l0 = lv.x, l1 = lv.y, l2 = lv.z, l3 = (lane == 31) ? 0.f : lv.w;
    // exclusive add-scan -> C
    float s1 = l0 + l1, s2 = s1 + l2, s3 = s2 + l3;
    float inc = s3;
#pragma unroll
    for (int off = 1; off < 32; off <<= 1) {
      float tt = __shfl_up(inc, off, 32);
      if (lane >= off) inc += tt;
    }
    float excl = inc - s3;
    float C0 = excl, C1 = excl + l0, C2 = excl + s1, C3 = excl + s2;
    // x = alpha + blank[t-1] - C, then inclusive lse-scan over u
    float x0 = a0 + bv.x - C0, x1 = a1 + bv.y - C1;
    float x2 = a2 + bv.z - C2, x3 = a3 + bv.w - C3;
    float y0 = x0, y1 = lse2(y0, x1), y2 = lse2(y1, x2), y3 = lse2(y2, x3);
    float linc = y3;
#pragma unroll
    for (int off = 1; off < 32; off <<= 1) {
      float tt = __shfl_up(linc, off, 32);
      if (lane >= off) linc = lse2(linc, tt);
    }
    float lex = __shfl_up(linc, 1, 32);       // exclusive across lanes
    if (lane == 0) lex = -INFINITY;
    a0 = C0 + lse2(lex, y0);
    a1 = C1 + lse2(lex, y1);
    a2 = C2 + lse2(lex, y2);
    a3 = C3 + lse2(lex, y3);
  }

  if ((Ul >> 2) == lane) {
    int k = Ul & 3;
    float av = (k == 0) ? a0 : (k == 1) ? a1 : (k == 2) ? a2 : a3;
    const float* bl = (const float*)bl4;
    out[b] = -(av + bl[(size_t)(Tl - 1) * U_ + Ul]);
  }
}

// ---------------------------------------------------------------------------
extern "C" void kernel_launch(void* const* d_in, const int* in_sizes, int n_in,
                              void* d_out, int out_size, void* d_ws, size_t ws_size,
                              hipStream_t stream) {
  const float* emissions   = (const float*)d_in[0];
  const float* predictions = (const float*)d_in[1];
  const int*   labels      = (const int*)d_in[2];
  const int*   in_len      = (const int*)d_in[3];
  const int*   lab_len     = (const int*)d_in[4];
  float* out = (float*)d_out;

  char* ws = (char*)d_ws;
  size_t off = 0;
  auto alloc = [&](size_t bytes) -> void* {
    void* ptr = ws + off;
    off += (bytes + 255) & ~(size_t)255;
    return ptr;
  };
  _Float16* expE = (_Float16*)alloc((size_t)B_ * T_ * V_ * 2);   // 64 MB
  _Float16* expP = (_Float16*)alloc((size_t)B_ * U_ * V_ * 2);   // 32 MB
  float* maxE  = (float*)alloc((size_t)B_ * T_ * 4);
  float* maxP  = (float*)alloc((size_t)B_ * U_ * 4);
  float* blank = (float*)alloc((size_t)B_ * T_ * U_ * 4);        // 4 MB
  float* lab   = (float*)alloc((size_t)B_ * T_ * U_ * 4);        // 4 MB
  (void)ws_size; (void)in_sizes; (void)n_in; (void)out_size;

  row_exp_kernel<<<B_ * T_, 256, 0, stream>>>(emissions, expE, maxE);
  row_exp_kernel<<<B_ * U_, 256, 0, stream>>>(predictions, expP, maxP);
  gemm_fused_kernel<<<dim3(T_ / 64, B_), 256, 0, stream>>>(
      expE, expP, maxE, maxP, emissions, predictions, labels, blank, lab);
  dp_kernel<<<B_, 32, 0, stream>>>(blank, lab, in_len, lab_len, out);
}
